// MultiHeadAttention_30614526886065
// MI455X (gfx1250) — matrix operand order repaired and checked by reading, not compiled, not run on hardware
//
#include <hip/hip_runtime.h>

#define SEQ    12
#define DMODEL 1024
#define NHEAD  16
#define HDIM   64
#define NBATCH 4096
#define NROWS  (NBATCH * SEQ)   // 49152

typedef __bf16 bf16;
typedef __attribute__((ext_vector_type(16))) __bf16 bf16x16;
typedef __attribute__((ext_vector_type(8)))  __bf16 bf16x8;
typedef __attribute__((ext_vector_type(8)))  float  v8f;
typedef __attribute__((ext_vector_type(4)))  float  v4f;
typedef __attribute__((ext_vector_type(4)))  unsigned int u32x4;
typedef __attribute__((ext_vector_type(8)))  int i32x8;
typedef __attribute__((ext_vector_type(4)))  int i32x4;

#if defined(__HIP_DEVICE_COMPILE__) && __has_builtin(__builtin_amdgcn_tensor_load_to_lds) && __has_builtin(__builtin_amdgcn_s_wait_tensorcnt)
#define HAVE_TDM 1
#endif

union Frag {
  bf16x16 v;
  bf16x8  h[2];
  bf16    e[16];
};

__device__ __forceinline__ bf16 f2bf(float f) {
  union { float f; unsigned u; } x; x.f = f;
  unsigned r = x.u + 0x7fffu + ((x.u >> 16) & 1u);   // round-to-nearest-even
  unsigned short hs = (unsigned short)(r >> 16);
  bf16 b;
  __builtin_memcpy(&b, &hs, sizeof(b));
  return b;
}

#ifdef HAVE_TDM
// ---------------------------------------------------------------------------
// Issue one TDM 2D tile load: 128 rows x 32 bf16, global row stride 1024 elems,
// into LDS at lds_off. Descriptor per CDNA5 ISA ch.8 (D# groups 0/1; 2D so
// groups 2/3 are zero).  EXEC is ignored by tensor ops; call from one wave.
// This toolchain uses the 6-arg builtin form (extra int32x8 group + cpol).
// ---------------------------------------------------------------------------
__device__ __forceinline__ void tdm_load_tile_128x32(const bf16* g, unsigned lds_off) {
  unsigned long long ga = (unsigned long long)(size_t)g;
  u32x4 g0 = {};
  g0[0] = 1u;                                    // count=1 (valid user D#)
  g0[1] = lds_off;                               // lds_addr (bytes)
  g0[2] = (unsigned)ga;                          // global_addr[31:0]
  g0[3] = ((unsigned)(ga >> 32) & 0x01ffffffu)   // global_addr[56:32]
          | (2u << 30);                          // type = 2 ("image")
  i32x8 g1 = {};
  g1[0] = 0x10000;                               // data_size = 1 -> 2 bytes
  g1[1] = (int)(32u << 16);                      // tensor_dim0 = 32 (lo16)
  g1[2] = (int)(128u << 16);                     // tensor_dim0 hi | tensor_dim1 = 128
  g1[3] = (int)(32u << 16);                      // tensor_dim1 hi | tile_dim0 = 32
  g1[4] = 128;                                   // tile_dim1 = 128, tile_dim2 = 0
  g1[5] = 1024;                                  // tensor_dim0_stride = 1024 elems
  g1[6] = 0;
  g1[7] = 0;
  i32x4 gz4 = {};
  i32x8 gz8 = {};
  __builtin_amdgcn_tensor_load_to_lds(g0, g1, gz4, gz4, gz8, 0);
}
#endif

// fallback / host-pass staging: cooperative 256-thread copy of both tiles
__device__ __forceinline__ void stage_copy_tiles(bf16 (*A)[32], const bf16* Ag,
                                                 bf16 (*B)[32], const bf16* Bg,
                                                 int t, int kk) {
  int r = t >> 1, c = (t & 1) * 16;
  const bf16* sa = Ag + (size_t)r * DMODEL + kk + c;
  *(bf16x8*)&A[r][c]     = *(const bf16x8*)sa;
  *(bf16x8*)&A[r][c + 8] = *(const bf16x8*)(sa + 8);
  const bf16* sb = Bg + (size_t)r * DMODEL + kk + c;
  *(bf16x8*)&B[r][c]     = *(const bf16x8*)sb;
  *(bf16x8*)&B[r][c + 8] = *(const bf16x8*)(sb + 8);
}

// ---------------------------------------------------------------------------
// K0: convert fp32 weights -> bf16, transposed: Wt[mat][n][k] = W[mat][k][n]
// ---------------------------------------------------------------------------
__global__ void k_prep_weights(const float* __restrict__ Wq, const float* __restrict__ Wk,
                               const float* __restrict__ Wv, const float* __restrict__ Wo,
                               bf16* __restrict__ Wt) {
  size_t id = (size_t)blockIdx.x * blockDim.x + threadIdx.x;
  if (id >= (size_t)4 * DMODEL * DMODEL) return;
  int mat = (int)(id >> 20);
  int rem = (int)(id & (DMODEL * DMODEL - 1));
  int k = rem >> 10;        // W row (input dim)
  int n = rem & 1023;       // W col (output dim), fastest -> coalesced read
  const float* W = (mat == 0) ? Wq : (mat == 1) ? Wk : (mat == 2) ? Wv : Wo;
  float v = W[(size_t)k * DMODEL + n];
  Wt[(size_t)mat * DMODEL * DMODEL + (size_t)n * DMODEL + k] = f2bf(v);
}

// ---------------------------------------------------------------------------
// K0b: build 16 row bitmasks from edge_index (int64 pairs)
// ---------------------------------------------------------------------------
__global__ void k_build_mask(const long long* __restrict__ edges, int nedges,
                             unsigned* __restrict__ maskArr) {
  if (blockIdx.x == 0 && threadIdx.x == 0) {
    unsigned m[16];
    for (int i = 0; i < 16; i++) m[i] = 0u;
    for (int e = 0; e < nedges; e++) {
      int s = (int)edges[2 * e];
      int d = (int)edges[2 * e + 1];
      if (s >= 0 && s < SEQ && d >= 0 && d < SEQ) m[s] |= (1u << d);
    }
    for (int i = 0; i < 16; i++) maskArr[i] = m[i];
  }
}

// ---------------------------------------------------------------------------
// K1: x fp32 -> bf16  (xb fits in L2: 96 MB < 192 MB)
// ---------------------------------------------------------------------------
__global__ void k_convert_x(const float* __restrict__ x, bf16* __restrict__ xb) {
  size_t i = ((size_t)blockIdx.x * blockDim.x + threadIdx.x) * 8;
  v4f a = *(const v4f*)(x + i);
  v4f b = *(const v4f*)(x + i + 4);
  bf16x8 o;
  o[0] = f2bf(a[0]); o[1] = f2bf(a[1]); o[2] = f2bf(a[2]); o[3] = f2bf(a[3]);
  o[4] = f2bf(b[0]); o[5] = f2bf(b[1]); o[6] = f2bf(b[2]); o[7] = f2bf(b[3]);
  *(bf16x8*)(xb + i) = o;
}

// ---------------------------------------------------------------------------
// K2: fused QKV GEMM with TDM double-buffered LDS pipeline.
// grid (384 Mtiles, 8 Ntiles, 3 mats), 256 thr = 8 waves.
// Wave 0 issues tensor_load_to_lds for the NEXT k-slice (A 128x32 from xb,
// B 128x32 from transposed Wt) while all waves run ds_load_b128 fragment
// reads + 8 WMMAs on the CURRENT buffer; s_wait_tensorcnt + barrier per step.
// ---------------------------------------------------------------------------
__global__ __launch_bounds__(256) void k_gemm_qkv(
    const bf16* __restrict__ xb, const bf16* __restrict__ Wt,
    const float* __restrict__ bq, const float* __restrict__ bk,
    const float* __restrict__ bv, bf16* __restrict__ qkv) {
  __shared__ bf16 ALds[2][128][32];
  __shared__ bf16 BLds[2][128][32];

  const int w    = threadIdx.x >> 5;
  const int lane = threadIdx.x & 31;
  const int half = lane >> 4;
  const int l16  = lane & 15;
  const int mt = blockIdx.x, nt = blockIdx.y, mat = blockIdx.z;

  const bf16* Ag = xb + (size_t)(mt * 128) * DMODEL;                         // [128][1024]
  const bf16* Bg = Wt + (size_t)mat * DMODEL * DMODEL
                      + (size_t)(nt * 128) * DMODEL;                         // [128][1024]
  const float* bias = (mat == 0) ? bq : (mat == 1) ? bk : bv;

  v8f acc[8];
  v8f vzero = {};
#pragma unroll
  for (int f = 0; f < 8; f++) acc[f] = vzero;

  // ---- prologue: stage k-slice 0 into buffer 0 ----
#ifdef HAVE_TDM
  if (w == 0) {
    tdm_load_tile_128x32(Ag, (unsigned)(size_t)&ALds[0][0][0]);
    tdm_load_tile_128x32(Bg, (unsigned)(size_t)&BLds[0][0][0]);
    __builtin_amdgcn_s_wait_tensorcnt(0);
  }
#else
  stage_copy_tiles(ALds[0], Ag, BLds[0], Bg, threadIdx.x, 0);
#endif
  __syncthreads();

  for (int ks = 0; ks < DMODEL / 32; ks++) {
    const int cur = ks & 1;
    const int nxt = cur ^ 1;

    // ---- kick DMA for next k-slice (its buffer was released by the barrier
    //      at the end of iteration ks-1) ----
    if (ks + 1 < DMODEL / 32) {
      const int kk = (ks + 1) * 32;
#ifdef HAVE_TDM
      if (w == 0) {
        tdm_load_tile_128x32(Ag + kk, (unsigned)(size_t)&ALds[nxt][0][0]);
        tdm_load_tile_128x32(Bg + kk, (unsigned)(size_t)&BLds[nxt][0][0]);
      }
#else
      stage_copy_tiles(ALds[nxt], Ag, BLds[nxt], Bg, threadIdx.x, kk);
#endif
    }

    // ---- consume current buffer: 1 A frag + 8 B frags, 8 WMMAs ----
    Frag a;
    a.h[0] = *(const bf16x8*)&ALds[cur][w * 16 + l16][half * 8];
    a.h[1] = *(const bf16x8*)&ALds[cur][w * 16 + l16][16 + half * 8];
#pragma unroll
    for (int f = 0; f < 8; f++) {
      Frag b;
      b.h[0] = *(const bf16x8*)&BLds[cur][f * 16 + l16][half * 8];
      b.h[1] = *(const bf16x8*)&BLds[cur][f * 16 + l16][16 + half * 8];
      acc[f] = __builtin_amdgcn_wmma_f32_16x16x32_bf16(
          false, a.v, false, b.v, (short)0, acc[f], false, false);
    }

    // ---- close the pipeline stage ----
#ifdef HAVE_TDM
    if (w == 0) __builtin_amdgcn_s_wait_tensorcnt(0);
#endif
    __syncthreads();
  }

  // ---- epilogue: bias + convert + store bf16 ----
  const size_t matsz = (size_t)NROWS * DMODEL;
#pragma unroll
  for (int f = 0; f < 8; f++) {
    int n = nt * 128 + f * 16 + l16;
    float bval = bias[n];
#pragma unroll
    for (int i = 0; i < 8; i++) {
      int r = mt * 128 + w * 16 + half * 8 + i;   // C/D layout: M = half*8 + i
      qkv[mat * matsz + (size_t)r * DMODEL + n] = f2bf(acc[f][i] + bval);
    }
  }
}

// ---------------------------------------------------------------------------
// K3: attention. One wave per (batch, head). 128 thr = 4 waves per block.
// scores = Q K^T (2 WMMAs), mask+softmax via shfl_xor over 16-lane halves,
// attn staged through LDS to reshape into WMMA A-layout, out = attn@V (4 WMMAs)
// ---------------------------------------------------------------------------
__global__ __launch_bounds__(128) void k_attention(
    const bf16* __restrict__ qkv, const unsigned* __restrict__ maskArr,
    bf16* __restrict__ ctx) {
  __shared__ bf16 attnLds[4][16][32];

  const int w    = threadIdx.x >> 5;
  const int lane = threadIdx.x & 31;
  const int half = lane >> 4;
  const int l16  = lane & 15;
  const int g = blockIdx.x * 4 + w;
  const int b = g >> 4;
  const int h = g & 15;

  const size_t matsz = (size_t)NROWS * DMODEL;
  const bf16* Qb = qkv + (size_t)(b * SEQ) * DMODEL + h * HDIM;
  const bf16* Kb = Qb + matsz;
  const bf16* Vb = Qb + 2 * matsz;

  const int rc = (l16 < SEQ) ? l16 : (SEQ - 1);   // clamp rows 12..15 (masked later)

  // ---- scores = Q K^T ----
  v8f s = {};
#pragma unroll
  for (int kk = 0; kk < HDIM; kk += 32) {
    Frag a, bk_;
    const bf16* qp = Qb + (size_t)rc * DMODEL + kk + half * 8;
    a.h[0] = *(const bf16x8*)(qp);
    a.h[1] = *(const bf16x8*)(qp + 16);
    const bf16* kp = Kb + (size_t)rc * DMODEL + kk + half * 8;
    bk_.h[0] = *(const bf16x8*)(kp);
    bk_.h[1] = *(const bf16x8*)(kp + 16);
    s = __builtin_amdgcn_wmma_f32_16x16x32_bf16(
        false, a.v, false, bk_.v, (short)0, s, false, false);
  }

  // ---- mask + softmax (per row M = half*8+i, cols spread over 16 lanes) ----
  unsigned mrow[8];
#pragma unroll
  for (int i = 0; i < 8; i++) mrow[i] = maskArr[half * 8 + i];

  const float scale = 0.125f;   // 1/sqrt(64)
  float attn[8];
#pragma unroll
  for (int i = 0; i < 8; i++) {
    bool on = ((mrow[i] >> l16) & 1u) && (l16 < SEQ);
    float v = on ? s[i] * scale : -1e30f;
    float m = v;
#pragma unroll
    for (int d = 1; d < 16; d <<= 1) m = fmaxf(m, __shfl_xor(m, d, 32));
    float e = on ? __expf(v - m) : 0.f;
    float sum = e;
#pragma unroll
    for (int d = 1; d < 16; d <<= 1) sum += __shfl_xor(sum, d, 32);
    float inv = (sum > 0.f) ? (1.f / sum) : 0.f;   // fully-masked rows -> 0
    attn[i] = e * inv;
  }

  // ---- stage attn into LDS, zero-padded K=16..31 ----
  bf16 z = f2bf(0.f);
  bf16x8 zv;
#pragma unroll
  for (int j = 0; j < 8; j++) zv[j] = z;
  *(bf16x8*)&attnLds[w][l16][16 + half * 8] = zv;
#pragma unroll
  for (int i = 0; i < 8; i++) attnLds[w][half * 8 + i][l16] = f2bf(attn[i]);
  __syncthreads();

  // ---- reload attn in WMMA A-layout ----
  Frag af;
  af.h[0] = *(const bf16x8*)&attnLds[w][l16][half * 16];
  af.h[1] = *(const bf16x8*)&attnLds[w][l16][half * 16 + 8];

  // ---- out = attn @ V  (4 WMMAs over the 64 hd columns) ----
#pragma unroll
  for (int ntile = 0; ntile < 4; ntile++) {
    Frag bv_;
#pragma unroll
    for (int j = 0; j < 16; j++) {
      int k  = half * 16 + j;
      int vr = (k < SEQ) ? k : (SEQ - 1);         // attn==0 there anyway
      bv_.e[j] = Vb[(size_t)vr * DMODEL + ntile * 16 + l16];
    }
    v8f c = {};
    v8f o = __builtin_amdgcn_wmma_f32_16x16x32_bf16(
        false, af.v, false, bv_.v, (short)0, c, false, false);
#pragma unroll
    for (int i = 0; i < 8; i++) {
      int r = half * 8 + i;
      if (r < SEQ)
        ctx[(size_t)(b * SEQ + r) * DMODEL + h * HDIM + ntile * 16 + l16] =
            f2bf(o[i]);
    }
  }
}

// ---------------------------------------------------------------------------
// K4: output projection + residual + LayerNorm, fused.
// One WG = 16 rows x full 1024 cols. 256 thr = 8 waves, wave w -> cols 128w..
// ---------------------------------------------------------------------------
__global__ __launch_bounds__(256) void k_gemm_out_ln(
    const bf16* __restrict__ ctx, const bf16* __restrict__ WtO,
    const float* __restrict__ bo, const float* __restrict__ x,
    const float* __restrict__ gamma, const float* __restrict__ beta,
    float* __restrict__ out) {
  __shared__ bf16  aLds[16][DMODEL + 8];   // +8 pad -> bank-staggered rows
  __shared__ float hbuf[16][DMODEL + 8];

  const int w    = threadIdx.x >> 5;
  const int lane = threadIdx.x & 31;
  const int half = lane >> 4;
  const int l16  = lane & 15;
  const int m0 = blockIdx.x * 16;

  // stage the 16x1024 bf16 A strip once
  for (int c = threadIdx.x; c < 16 * 128; c += 256) {
    int r = c >> 7, c8 = (c & 127) * 8;
    *(bf16x8*)&aLds[r][c8] = *(const bf16x8*)(ctx + (size_t)(m0 + r) * DMODEL + c8);
  }
  __syncthreads();

  v8f acc[8];
  v8f vzero = {};
#pragma unroll
  for (int f = 0; f < 8; f++) acc[f] = vzero;

  for (int kk = 0; kk < DMODEL; kk += 32) {
    Frag a;
    a.h[0] = *(const bf16x8*)&aLds[l16][kk + half * 8];
    a.h[1] = *(const bf16x8*)&aLds[l16][kk + 16 + half * 8];
#pragma unroll
    for (int f = 0; f < 8; f++) {
      int n = w * 128 + f * 16 + l16;
      const bf16* brow = WtO + (size_t)n * DMODEL + kk + half * 8;
      __builtin_prefetch(brow + 32, 0, 2);
      Frag b;
      b.h[0] = *(const bf16x8*)(brow);
      b.h[1] = *(const bf16x8*)(brow + 16);
      acc[f] = __builtin_amdgcn_wmma_f32_16x16x32_bf16(
          false, a.v, false, b.v, (short)0, acc[f], false, false);
    }
  }

  // epilogue: bias + residual into LDS
#pragma unroll
  for (int f = 0; f < 8; f++) {
    int n = w * 128 + f * 16 + l16;
    float bb = bo[n];
#pragma unroll
    for (int i = 0; i < 8; i++) {
      int r = half * 8 + i;
      hbuf[r][n] = acc[f][i] + bb + x[(size_t)(m0 + r) * DMODEL + n];
    }
  }
  __syncthreads();

  // LayerNorm: wave w handles rows 2w, 2w+1; full-wave shuffle reduction
  for (int rr = 0; rr < 2; rr++) {
    int r = w * 2 + rr;
    float sum = 0.f, ss = 0.f;
    for (int c = lane; c < DMODEL; c += 32) {
      float v = hbuf[r][c];
      sum += v;
      ss  += v * v;
    }
#pragma unroll
    for (int d = 1; d < 32; d <<= 1) {
      sum += __shfl_xor(sum, d, 32);
      ss  += __shfl_xor(ss,  d, 32);
    }
    float mu   = sum * (1.f / DMODEL);
    float var  = ss * (1.f / DMODEL) - mu * mu;
    float rstd = rsqrtf(var + 1e-5f);
    for (int c = lane; c < DMODEL; c += 32) {
      float v = hbuf[r][c];
      out[(size_t)(m0 + r) * DMODEL + c] = (v - mu) * rstd * gamma[c] + beta[c];
    }
  }
}

// ---------------------------------------------------------------------------
extern "C" void kernel_launch(void* const* d_in, const int* in_sizes, int n_in,
                              void* d_out, int out_size, void* d_ws, size_t ws_size,
                              hipStream_t stream) {
  const float*     x     = (const float*)d_in[0];
  const long long* edges = (const long long*)d_in[1];
  const float* Wq = (const float*)d_in[2];  const float* bq = (const float*)d_in[3];
  const float* Wk = (const float*)d_in[4];  const float* bk = (const float*)d_in[5];
  const float* Wv = (const float*)d_in[6];  const float* bv = (const float*)d_in[7];
  const float* Wo = (const float*)d_in[8];  const float* bo = (const float*)d_in[9];
  const float* gamma = (const float*)d_in[10];
  const float* beta  = (const float*)d_in[11];
  float* out = (float*)d_out;

  char* ws = (char*)d_ws;
  const size_t wsz   = 4ull * DMODEL * DMODEL * sizeof(bf16);   // 8 MB
  const size_t matsz = (size_t)NROWS * DMODEL * sizeof(bf16);   // 96 MB
  bf16*     Wt      = (bf16*)ws;
  bf16*     xb      = (bf16*)(ws + wsz);
  bf16*     qkv     = (bf16*)(ws + wsz + matsz);
  bf16*     ctx     = (bf16*)(ws + wsz + matsz + 3 * matsz);
  unsigned* maskArr = (unsigned*)(ws + wsz + matsz + 4 * matsz);

  const int nedges = in_sizes[1] / 2;

  k_prep_weights<<<(4 * DMODEL * DMODEL) / 256, 256, 0, stream>>>(Wq, Wk, Wv, Wo, Wt);
  k_build_mask<<<1, 32, 0, stream>>>(edges, nedges, maskArr);
  k_convert_x<<<(NROWS * DMODEL) / (256 * 8), 256, 0, stream>>>(x, xb);

  dim3 g2(NROWS / 128, DMODEL / 128, 3);
  k_gemm_qkv<<<g2, 256, 0, stream>>>(xb, Wt, bq, bk, bv, qkv);

  k_attention<<<(NBATCH * NHEAD) / 4, 128, 0, stream>>>(qkv, maskArr, ctx);

  k_gemm_out_ln<<<NROWS / 16, 256, 0, stream>>>(
      ctx, Wt + 3ull * DMODEL * DMODEL, bo, x, gamma, beta, out);
}
